// MultiHeadAttention_62397284876972
// MI455X (gfx1250) — matrix operand order repaired and checked by reading, not compiled, not run on hardware
//
#include <hip/hip_runtime.h>

#define KD   1024
#define NH   16
#define HS   64
#define BATCH 4
#define SEQ  2048
#define MTOT (BATCH*SEQ)   // 8192

typedef __attribute__((ext_vector_type(16))) __bf16 v16bf;
typedef __attribute__((ext_vector_type(8)))  __bf16 v8bf;
typedef __attribute__((ext_vector_type(8)))  float  v8f;
typedef unsigned int u32x4 __attribute__((ext_vector_type(4)));
typedef int          i32x4 __attribute__((ext_vector_type(4)));
typedef int          i32x8 __attribute__((ext_vector_type(8)));

// ---- vectorized fragment loaders (wave32, 16x16x32 bf16 WMMA) ----
// A matrix 16x32: lane (half = lane>>4) holds row M = lane&15,
// element i -> K = (i/8)*16 + half*8 + (i%8)  => two 16B runs
__device__ __forceinline__ v16bf fragA(const __bf16* __restrict__ row, int half) {
  v8bf lo = *(const v8bf*)(row + half * 8);
  v8bf hi = *(const v8bf*)(row + half * 8 + 16);
  return __builtin_shufflevector(lo, hi, 0,1,2,3,4,5,6,7,8,9,10,11,12,13,14,15);
}
// B matrix 32x16: lane holds column N = lane&15, element i -> K = (i/8)*16 + half*8 + (i%8)
// => two 16B runs (two 16B vector loads)
__device__ __forceinline__ v16bf fragB(const __bf16* __restrict__ col, int half) {
  v8bf lo = *(const v8bf*)(col + half * 8);
  v8bf hi = *(const v8bf*)(col + half * 8 + 16);
  return __builtin_shufflevector(lo, hi, 0,1,2,3,4,5,6,7,8,9,10,11,12,13,14,15);
}

#define WMMA_BF16(a, b, c) \
  __builtin_amdgcn_wmma_f32_16x16x32_bf16(false, (a), false, (b), (short)0, (c), false, false)

// ---- Tensor Data Mover: 2D tile (bf16) global -> LDS ----
// D# per CDNA5 ISA sec.8. Groups 2/3 zero (<=2D tensor).
__device__ __forceinline__ void tdm_load_2d(unsigned lds_off, const void* gptr,
                                            unsigned tile_d0, unsigned tile_d1,
                                            unsigned row_len_elems) {
  unsigned long long ga = (unsigned long long)(size_t)gptr;
  u32x4 g0 = { 1u,                         // count=1, user descriptor
               lds_off,                    // lds_addr (bytes)
               (unsigned)ga,               // global_addr[31:0]
               (unsigned)((ga >> 32) & 0x01FFFFFFu) | (2u << 30) };  // [56:32] | type=2

  const unsigned td0 = row_len_elems;      // tensor_dim0 (elems)
  const unsigned td1 = 0x100000u;          // tensor_dim1: large, tiles stay in-bounds
  i32x8 g1 = { (int)0x00010000u,                              // data_size=1 (2 bytes)
               (int)((td0 & 0xFFFFu) << 16),                  // tensor_dim0 lo16 @ [63:48]
               (int)((td0 >> 16) | ((td1 & 0xFFFFu) << 16)),  // dim0 hi | dim1 lo
               (int)((td1 >> 16) | (tile_d0 << 16)),          // dim1 hi | tile_dim0
               (int)tile_d1,                                  // tile_dim1 (tile_dim2=0)
               (int)row_len_elems,                            // tensor_dim0_stride lo32
               0, 0 };
  i32x4 z4 = {0, 0, 0, 0};
#if defined(__clang_major__) && (__clang_major__ >= 23)
  i32x8 z8 = {0, 0, 0, 0, 0, 0, 0, 0};
  __builtin_amdgcn_tensor_load_to_lds(g0, g1, z4, z4, z8, 0);
#else
  __builtin_amdgcn_tensor_load_to_lds(g0, g1, z4, z4, 0);
#endif
}

// ============================================================
// Kernel 0: one-time fp32 -> bf16 conversion (x and the 4 weights).
// ============================================================
__global__ __launch_bounds__(256)
void cvt_kernel(const float* __restrict__ s, __bf16* __restrict__ d, int n) {
  int i = (blockIdx.x * 256 + threadIdx.x) * 8;
  if (i >= n) return;
  float4 a = *(const float4*)(s + i);
  float4 b = *(const float4*)(s + i + 4);
  v8bf o;
  o[0] = (__bf16)a.x; o[1] = (__bf16)a.y; o[2] = (__bf16)a.z; o[3] = (__bf16)a.w;
  o[4] = (__bf16)b.x; o[5] = (__bf16)b.y; o[6] = (__bf16)b.z; o[7] = (__bf16)b.w;
  *(v8bf*)(d + i) = o;
}

// ============================================================
// Kernel 1: fused QKV projection, 32x64 tile per wave (8 WMMA / k-step).
// Q,K written [b,h,t,hs]; V written TRANSPOSED [b,h,hs,t].
// ============================================================
__global__ __launch_bounds__(256)
void qkv_proj_kernel(const __bf16* __restrict__ xb,
                     const __bf16* __restrict__ Wq,
                     const __bf16* __restrict__ Wk,
                     const __bf16* __restrict__ Wv,
                     __bf16* __restrict__ Qb,
                     __bf16* __restrict__ Kb,
                     __bf16* __restrict__ Vt) {
  const int wave = threadIdx.x >> 5;
  const int lane = threadIdx.x & 31;
  const int half = lane >> 4;
  const int lid  = lane & 15;

  const int MT = MTOT / 32;  // 256
  const int NT = KD / 64;    // 16
  int gw  = blockIdx.x * 8 + wave;
  int mat = gw / (MT * NT);
  int rem = gw % (MT * NT);
  int mt = rem / NT;
  int nt = rem % NT;       // == head index (64-wide n-tile == one head)

  const __bf16* W   = (mat == 0) ? Wq : ((mat == 1) ? Wk : Wv);
  __bf16*       Out = (mat == 0) ? Qb : ((mat == 1) ? Kb : Vt);

  const int m0 = mt * 32, n0 = nt * 64;
  const __bf16* arow0 = xb + (size_t)(m0 + lid) * KD;
  const __bf16* arow1 = arow0 + (size_t)16 * KD;
  const __bf16* brow0 = W + (size_t)(n0 + lid) * KD;

  v8f c[2][4] = {};
  for (int kk = 0; kk < KD; kk += 32) {
    v16bf a0 = fragA(arow0 + kk, half);
    v16bf a1 = fragA(arow1 + kk, half);
#pragma unroll
    for (int j = 0; j < 4; ++j) {
      v16bf b = fragB(brow0 + (size_t)j * 16 * KD + kk, half);
      c[0][j] = WMMA_BF16(a0, b, c[0][j]);
      c[1][j] = WMMA_BF16(a1, b, c[1][j]);
    }
  }

#pragma unroll
  for (int i = 0; i < 2; ++i) {
#pragma unroll
    for (int r = 0; r < 8; ++r) {
      int mrow = m0 + i * 16 + r + 8 * half;
      int b_ = mrow / SEQ, t_ = mrow % SEQ;
      if (mat == 2) {
        size_t base = (((size_t)b_ * NH + nt) * HS) * SEQ + t_;   // [b,h,d,t]
#pragma unroll
        for (int j = 0; j < 4; ++j)
          Out[base + (size_t)(j * 16 + lid) * SEQ] = (__bf16)c[i][j][r];
      } else {
        size_t base = (((size_t)b_ * NH + nt) * SEQ + t_) * HS;   // [b,h,t,d]
#pragma unroll
        for (int j = 0; j < 4; ++j)
          Out[base + j * 16 + lid] = (__bf16)c[i][j][r];
      }
    }
  }
}

// ============================================================
// Kernel 2: causal flash-attention, one WAVE per block (32 threads).
// 32 queries per wave; 32-key blocks TDM-staged into LDS with
// double buffering (TENSORcnt pipelining); 16 WMMAs per block iter.
// ============================================================
__global__ __launch_bounds__(32)
void attn_kernel(const __bf16* __restrict__ Qb,
                 const __bf16* __restrict__ Kb,
                 const __bf16* __restrict__ Vt,
                 __bf16* __restrict__ Ob) {
  __shared__ __align__(16) __bf16 kst[2][32 * HS];  // [key][d]   4KB x2
  __shared__ __align__(16) __bf16 vst[2][HS * 32];  // [d][key]   4KB x2
  __shared__ __align__(16) __bf16 pbuf[32 * 32];    // P staging  2KB

  const int lane = threadIdx.x & 31;
  const int half = lane >> 4;
  const int lid  = lane & 15;

  const int QT = SEQ / 32;  // 64 query blocks per (b,h)
  int gw = blockIdx.x;
  int bh = gw / QT;
  int qt = gw % QT;
  int q0 = qt * 32;
  int b_ = bh / NH, h_ = bh % NH;

  const __bf16* Qm = Qb + (size_t)bh * SEQ * HS;
  const __bf16* Km = Kb + (size_t)bh * SEQ * HS;
  const __bf16* Vm = Vt + (size_t)bh * HS * SEQ;   // [d][t]

  unsigned koff[2], voff[2];
  koff[0] = (unsigned)(size_t)(void*)&kst[0][0];
  koff[1] = (unsigned)(size_t)(void*)&kst[1][0];
  voff[0] = (unsigned)(size_t)(void*)&vst[0][0];
  voff[1] = (unsigned)(size_t)(void*)&vst[1][0];

  // Q: 32x64 -> 2 q-tiles x 2 A fragments, loaded once
  v16bf qa[2][2];
#pragma unroll
  for (int g = 0; g < 2; ++g) {
    const __bf16* qrow = Qm + (size_t)(q0 + g * 16 + lid) * HS;
    qa[g][0] = fragA(qrow, half);
    qa[g][1] = fragA(qrow + 32, half);
  }

  v8f o[2][4] = {};
  float mrow[2][8], lrow[2][8];
#pragma unroll
  for (int g = 0; g < 2; ++g)
#pragma unroll
    for (int r = 0; r < 8; ++r) { mrow[g][r] = -__builtin_huge_valf(); lrow[g][r] = 0.f; }

  const float sc = 0.03125f * 1.44269504088896f;  // 1/sqrt(1024) * log2(e)

  const int nblocks = qt + 1;  // (q0+32)/32

  // --- prologue: TDM-stage key block 0 into buffer 0 ---
  tdm_load_2d(koff[0], Km, HS, 32, HS);           // K tile: 32 rows x 64 elems
  tdm_load_2d(voff[0], Vm, 32, HS, SEQ);          // V^T tile: 64 rows x 32 elems

  int buf = 0;
  for (int ib = 0; ib < nblocks; ++ib) {
    int jb = ib * 32;

    // --- issue next block's TDM loads, then wait for current block ---
    if (ib + 1 < nblocks) {
      int jn = jb + 32;
      tdm_load_2d(koff[buf ^ 1], Km + (size_t)jn * HS, HS, 32, HS);
      tdm_load_2d(voff[buf ^ 1], Vm + jn, 32, HS, SEQ);
      __builtin_amdgcn_s_wait_tensorcnt(2);   // in-order: current 2 loads done
    } else {
      __builtin_amdgcn_s_wait_tensorcnt(0);
    }
    asm volatile("" ::: "memory");

    const __bf16* kl = &kst[buf][0];
    const __bf16* vl = &vst[buf][0];

    // ---- K fragments from LDS (shared by both q-tiles) ----
    v16bf kb00 = fragB(kl + (size_t)lid * HS, half);
    v16bf kb01 = fragB(kl + (size_t)lid * HS + 32, half);
    v16bf kb10 = fragB(kl + (size_t)(16 + lid) * HS, half);
    v16bf kb11 = fragB(kl + (size_t)(16 + lid) * HS + 32, half);

#pragma unroll
    for (int g = 0; g < 2; ++g) {
      // ---- S = Q K^T : 16x32 as two 16x16 C tiles ----
      v8f s0 = {}, s1 = {};
      s0 = WMMA_BF16(qa[g][0], kb00, s0);
      s0 = WMMA_BF16(qa[g][1], kb01, s0);
      s1 = WMMA_BF16(qa[g][0], kb10, s1);
      s1 = WMMA_BF16(qa[g][1], kb11, s1);

      // ---- scale + causal mask + online softmax ----
#pragma unroll
      for (int r = 0; r < 8; ++r) {
        int qrow_i = q0 + g * 16 + r + 8 * half;
        float v0 = s0[r] * sc, v1 = s1[r] * sc;
        if (jb + lid      > qrow_i) v0 = -__builtin_huge_valf();
        if (jb + 16 + lid > qrow_i) v1 = -__builtin_huge_valf();

        float mx = fmaxf(v0, v1);
#pragma unroll
        for (int d = 1; d < 16; d <<= 1) mx = fmaxf(mx, __shfl_xor(mx, d, 32));
        float mnew  = fmaxf(mrow[g][r], mx);
        float alpha = exp2f(mrow[g][r] - mnew);
        float p0 = exp2f(v0 - mnew);
        float p1 = exp2f(v1 - mnew);
        float sum = p0 + p1;
#pragma unroll
        for (int d = 1; d < 16; d <<= 1) sum += __shfl_xor(sum, d, 32);

        lrow[g][r] = lrow[g][r] * alpha + sum;
        mrow[g][r] = mnew;
        o[g][0][r] *= alpha; o[g][1][r] *= alpha;
        o[g][2][r] *= alpha; o[g][3][r] *= alpha;

        int prow_i = g * 16 + r + 8 * half;
        pbuf[prow_i * 32 + lid]      = (__bf16)p0;
        pbuf[prow_i * 32 + 16 + lid] = (__bf16)p1;
      }
    }

    // same-wave DS ordering fence (DS ops execute in issue order within a wave)
    asm volatile("s_wait_dscnt 0" ::: "memory");

    // ---- P (32x32) as two bf16 A fragments (vector LDS reads) ----
    v16bf pa[2];
#pragma unroll
    for (int g = 0; g < 2; ++g) {
      const __bf16* prow = pbuf + (g * 16 + lid) * 32;
      v8bf lo = *(const v8bf*)(prow + half * 8);
      v8bf hi = *(const v8bf*)(prow + 16 + half * 8);
      pa[g] = __builtin_shufflevector(lo, hi, 0,1,2,3,4,5,6,7,8,9,10,11,12,13,14,15);
    }
    asm volatile("s_wait_dscnt 0" ::: "memory");  // keep next stores after these loads

    // ---- O += P @ V : V^T LDS rows contiguous along keys ----
#pragma unroll
    for (int nt = 0; nt < 4; ++nt) {
      v16bf vb = fragB(vl + (size_t)(nt * 16 + lid) * 32, half);
      o[0][nt] = WMMA_BF16(pa[0], vb, o[0][nt]);
      o[1][nt] = WMMA_BF16(pa[1], vb, o[1][nt]);
    }

    buf ^= 1;
  }

  // ---- normalize and write O in [b, t, k] layout (k = h*64 + d) ----
#pragma unroll
  for (int g = 0; g < 2; ++g) {
#pragma unroll
    for (int r = 0; r < 8; ++r) {
      float inv = 1.0f / lrow[g][r];
      int trow = q0 + g * 16 + r + 8 * half;
      size_t base = ((size_t)b_ * SEQ + trow) * KD + (size_t)h_ * HS;
#pragma unroll
      for (int nt = 0; nt < 4; ++nt)
        Ob[base + nt * 16 + lid] = (__bf16)(o[g][nt][r] * inv);
    }
  }
}

// ============================================================
// Kernel 3: output projection  out = O @ Wo^T, 32x64 tile per wave, fp32 out.
// ============================================================
__global__ __launch_bounds__(256)
void out_proj_kernel(const __bf16* __restrict__ Ob,
                     const __bf16* __restrict__ Wo,
                     float* __restrict__ Y) {
  const int wave = threadIdx.x >> 5;
  const int lane = threadIdx.x & 31;
  const int half = lane >> 4;
  const int lid  = lane & 15;

  const int NT = KD / 64;  // 16
  int gw = blockIdx.x * 8 + wave;
  int mt = gw / NT;
  int nt = gw % NT;
  const int m0 = mt * 32, n0 = nt * 64;

  const __bf16* arow0 = Ob + (size_t)(m0 + lid) * KD;
  const __bf16* arow1 = arow0 + (size_t)16 * KD;
  const __bf16* brow0 = Wo + (size_t)(n0 + lid) * KD;

  v8f c[2][4] = {};
  for (int kk = 0; kk < KD; kk += 32) {
    v16bf a0 = fragA(arow0 + kk, half);
    v16bf a1 = fragA(arow1 + kk, half);
#pragma unroll
    for (int j = 0; j < 4; ++j) {
      v16bf b = fragB(brow0 + (size_t)j * 16 * KD + kk, half);
      c[0][j] = WMMA_BF16(a0, b, c[0][j]);
      c[1][j] = WMMA_BF16(a1, b, c[1][j]);
    }
  }

#pragma unroll
  for (int i = 0; i < 2; ++i) {
#pragma unroll
    for (int r = 0; r < 8; ++r) {
      int mrow = m0 + i * 16 + r + 8 * half;
#pragma unroll
      for (int j = 0; j < 4; ++j)
        Y[(size_t)mrow * KD + n0 + j * 16 + lid] = c[i][j][r];
    }
  }
}

// ============================================================
extern "C" void kernel_launch(void* const* d_in, const int* in_sizes, int n_in,
                              void* d_out, int out_size, void* d_ws, size_t ws_size,
                              hipStream_t stream) {
  const float* x  = (const float*)d_in[0];
  const float* Wq = (const float*)d_in[1];
  const float* Wk = (const float*)d_in[2];
  const float* Wv = (const float*)d_in[3];
  const float* Wo = (const float*)d_in[4];
  float* out = (float*)d_out;

  const size_t xe = (size_t)MTOT * KD;  // 8,388,608
  const size_t we = (size_t)KD * KD;    // 1,048,576

  __bf16* xb  = (__bf16*)d_ws;
  __bf16* wqb = xb  + xe;
  __bf16* wkb = wqb + we;
  __bf16* wvb = wkb + we;
  __bf16* wob = wvb + we;
  __bf16* qb  = wob + we;
  __bf16* kb  = qb  + xe;
  __bf16* vt  = kb  + xe;   // V transposed [b,h,hs,t]
  __bf16* ob  = vt  + xe;   // [b, t, k] bf16

  // --- stage 0: fp32 -> bf16 conversions ---
  cvt_kernel<<<(int)(xe / 2048), 256, 0, stream>>>(x,  xb,  (int)xe);
  cvt_kernel<<<(int)(we / 2048), 256, 0, stream>>>(Wq, wqb, (int)we);
  cvt_kernel<<<(int)(we / 2048), 256, 0, stream>>>(Wk, wkb, (int)we);
  cvt_kernel<<<(int)(we / 2048), 256, 0, stream>>>(Wv, wvb, (int)we);
  cvt_kernel<<<(int)(we / 2048), 256, 0, stream>>>(Wo, wob, (int)we);

  // --- stage 1: QKV projections (3 * 256 * 16 tiles, 8 waves/block) ---
  const int qkv_blocks = 3 * (MTOT / 32) * (KD / 64) / 8;  // 1536
  qkv_proj_kernel<<<qkv_blocks, 256, 0, stream>>>(xb, wqb, wkb, wvb, qb, kb, vt);

  // --- stage 2: attention (one wave per block, TDM double-buffered K/V) ---
  const int attn_blocks = BATCH * NH * (SEQ / 32);         // 4096
  attn_kernel<<<attn_blocks, 32, 0, stream>>>(qb, kb, vt, ob);

  // --- stage 3: output projection ---
  const int op_blocks = (MTOT / 32) * (KD / 64) / 8;       // 512
  out_proj_kernel<<<op_blocks, 256, 0, stream>>>(ob, wob, out);
}
